// Attention_17102559773277
// MI455X (gfx1250) — compile-verified, matrix operand order checked by reading, not run on hardware
//
#include <hip/hip_runtime.h>
#include <hip/hip_bf16.h>

// GPT-2 attention block for MI455X (gfx1250, wave32, WMMA f16 16x16x32).
// B=2, S=2048, NX=1024, H=16, D=64.
// Register macro-tiling (GEMM: 32x64/wave, attention: 32 queries/wave)
// to lift arithmetic intensity from ~8 to ~22+ FLOP/byte of L2 traffic.

typedef _Float16 half_t;
typedef __attribute__((ext_vector_type(16))) _Float16 v16h;
typedef __attribute__((ext_vector_type(8)))  _Float16 v8h;
typedef __attribute__((ext_vector_type(8)))  float    v8f;

union Frag16 { v16h v; v8h h[2]; };

// A/B fragment for 16x16x32 f16 WMMA: lane (row|col)=l&15, half=l>>4 holds
// K = half*8..+7 in elements 0..7 and K = 16+half*8..+7 in elements 8..15
// -> two contiguous 16-byte loads when K is the fast axis.
__device__ __forceinline__ v16h load_frag(const half_t* p0, const half_t* p1) {
  Frag16 u;
  u.h[0] = *(const v8h*)p0;
  u.h[1] = *(const v8h*)p1;
  return u.v;
}

__device__ __forceinline__ v8f wmma_f16(v16h a, v16h b, v8f c) {
  return __builtin_amdgcn_wmma_f32_16x16x32_f16(
      /*neg_a=*/false, a, /*neg_b=*/false, b,
      /*c_mod=*/(short)0, c, /*reuse_a=*/false, /*reuse_b=*/false);
}

// ---------------------------------------------------------------- conversions
__global__ void __launch_bounds__(256) cvt_kernel(const float* __restrict__ in,
                                                  half_t* __restrict__ out, int n) {
  int i = blockIdx.x * blockDim.x + threadIdx.x;
  if (i < n) out[i] = (half_t)in[i];
}

// out[c][r] = in[r][c]  (transpose + f32->f16), out is [cols][rows]
__global__ void __launch_bounds__(256) cvtT_kernel(const float* __restrict__ in,
                                                   half_t* __restrict__ out,
                                                   int rows, int cols) {
  int i = blockIdx.x * blockDim.x + threadIdx.x;
  if (i < rows * cols) {
    int c = i / rows;
    int r = i - c * rows;
    out[i] = (half_t)in[r * cols + c];
  }
}

// ------------------------------------------------------------------ QKV GEMM
// C[4096][3072] = Xh @ w_attn^T + b_attn. One wave per 32x64 macro-tile:
// per k-step, 2 A frags + 4 B frags feed 8 WMMAs (22 FLOP/byte).
__global__ void __launch_bounds__(256) gemm_qkv_kernel(
    const half_t* __restrict__ X,    // [4096][1024]
    const half_t* __restrict__ WT,   // [3072][1024] (w_attn transposed)
    const float*  __restrict__ bias, // [3072]
    half_t* __restrict__ qk,         // [4096][2048]
    half_t* __restrict__ vT)         // [B*H*64][2048] = v[b][h][d][s]
{
  const int lane = threadIdx.x & 31;
  const int nlo  = lane & 15;
  const int hi   = lane >> 4;
  const int wave = blockIdx.x * 8 + (threadIdx.x >> 5);
  const int NTile = 3072 / 64;              // 48 macro-col tiles
  const int mt = wave / NTile;              // 0..127
  const int nt = wave - mt * NTile;

  const half_t* aB0 = X + (size_t)(mt * 32 + nlo) * 1024 + hi * 8;
  const half_t* aB1 = aB0 + (size_t)16 * 1024;
  const half_t* bB[4];
#pragma unroll
  for (int j = 0; j < 4; ++j)
    bB[j] = WT + (size_t)(nt * 64 + j * 16 + nlo) * 1024 + hi * 8;

  v8f acc[2][4] = {};
  for (int kk = 0; kk < 1024; kk += 32) {
    __builtin_prefetch(aB0 + kk + 256, 0, 1);
    __builtin_prefetch(bB[0] + kk + 256, 0, 1);
    v16h a0 = load_frag(aB0 + kk, aB0 + kk + 16);
    v16h a1 = load_frag(aB1 + kk, aB1 + kk + 16);
#pragma unroll
    for (int j = 0; j < 4; ++j) {
      v16h bf = load_frag(bB[j] + kk, bB[j] + kk + 16);
      acc[0][j] = wmma_f16(a0, bf, acc[0][j]);
      acc[1][j] = wmma_f16(a1, bf, acc[1][j]);
    }
  }

#pragma unroll
  for (int j = 0; j < 4; ++j) {
    const int n  = nt * 64 + j * 16 + nlo;
    const float bn = bias[n];
#pragma unroll
    for (int i = 0; i < 2; ++i) {
#pragma unroll
      for (int r = 0; r < 8; ++r) {
        const int m = mt * 32 + i * 16 + r + hi * 8;
        half_t hv = (half_t)(acc[i][j][r] + bn);
        if (n < 2048) {                     // uniform per 16-col subtile
          qk[(size_t)m * 2048 + n] = hv;
        } else {                            // v -> transposed [b][h][d][s]
          const int c  = n - 2048;
          const int bb = m >> 11, s = m & 2047;
          const int hh = c >> 6,  d = c & 63;
          vT[(size_t)(((bb * 16 + hh) << 6) + d) * 2048 + s] = hv;
        }
      }
    }
  }
}

// ------------------------------------------------------------ flash attention
// One wave per (b, h, 32-query group): 2 q-tiles share every K/V fragment.
// Online softmax over 32-key chunks up to the causal diagonal.
__global__ void __launch_bounds__(256) attn_kernel(
    const half_t* __restrict__ qk,  // [4096][2048]: cols 0..1023 q, 1024.. k
    const half_t* __restrict__ vT,  // [B*H*64][2048]
    half_t* __restrict__ ah)        // [4096][1024] merged-heads output (f16)
{
  __shared__ half_t ldsP[8][16 * 32];       // 1 KB per wave (P transpose bounce)
  const int lane = threadIdx.x & 31;
  const int nlo  = lane & 15;
  const int hi   = lane >> 4;
  const int wib  = threadIdx.x >> 5;
  const int wave = blockIdx.x * 8 + wib;    // 2048 waves total
  const int qgrp = wave & 63;               // S/32 = 64 query groups
  const int bh   = wave >> 6;
  const int b    = bh >> 4;
  const int h    = bh & 15;
  const int qbase = qgrp * 32;

  const half_t* qB[2];
#pragma unroll
  for (int qt = 0; qt < 2; ++qt)
    qB[qt] = qk + (size_t)(b * 2048 + qbase + qt * 16 + nlo) * 2048 + h * 64 + hi * 8;
  const half_t* kCol  = qk + 1024 + h * 64 + hi * 8;
  const half_t* vBase = vT + (size_t)((b * 16 + h) * 64) * 2048 + hi * 8;
  half_t* myP = &ldsP[wib][0];

  v8f o[2][4] = {};                         // 32 q rows x 64 d cols
  float Mrow[2][8], Lrow[2][8];
#pragma unroll
  for (int qt = 0; qt < 2; ++qt)
#pragma unroll
    for (int r = 0; r < 8; ++r) { Mrow[qt][r] = -3.0e38f; Lrow[qt][r] = 0.0f; }

  for (int kb = 0; kb <= qbase + 31; kb += 32) {
    // load K/V fragments once; both q-tiles reuse them
    v16h kf[2][2], vf[4];
#pragma unroll
    for (int d2 = 0; d2 < 2; ++d2)
#pragma unroll
      for (int sub = 0; sub < 2; ++sub) {
        const half_t* kp = kCol + (size_t)(b * 2048 + kb + sub * 16 + nlo) * 2048 + d2 * 32;
        kf[d2][sub] = load_frag(kp, kp + 16);
      }
#pragma unroll
    for (int j = 0; j < 4; ++j) {
      const half_t* vp = vBase + (size_t)(j * 16 + nlo) * 2048 + kb;
      vf[j] = load_frag(vp, vp + 16);
    }

#pragma unroll
    for (int qt = 0; qt < 2; ++qt) {
      const int qtb = qbase + qt * 16;
      if (kb > qtb + 15) continue;          // wave-uniform causal skip (qt==0)

      v8f s0 = {}, s1 = {};
#pragma unroll
      for (int d2 = 0; d2 < 2; ++d2) {
        v16h aq = load_frag(qB[qt] + d2 * 32, qB[qt] + d2 * 32 + 16);
        s0 = wmma_f16(aq, kf[d2][0], s0);
        s1 = wmma_f16(aq, kf[d2][1], s1);
      }

      // online softmax; rows live across the 16 lanes of each half-wave
#pragma unroll
      for (int r = 0; r < 8; ++r) {
        const int qrow = qtb + r + hi * 8;
        float v0 = s0[r] * 0.125f;          // 1/sqrt(64)
        float v1 = s1[r] * 0.125f;
        if (kb + nlo      > qrow) v0 = -1.0e30f;   // causal mask
        if (kb + 16 + nlo > qrow) v1 = -1.0e30f;
        float mx = fmaxf(v0, v1);
#pragma unroll
        for (int off = 1; off < 16; off <<= 1)
          mx = fmaxf(mx, __shfl_xor(mx, off, 16));
        const float Mnew = fmaxf(Mrow[qt][r], mx);
        const float sc = __expf(Mrow[qt][r] - Mnew);
        const float p0 = __expf(v0 - Mnew);
        const float p1 = __expf(v1 - Mnew);
        float rs = p0 + p1;
#pragma unroll
        for (int off = 1; off < 16; off <<= 1)
          rs += __shfl_xor(rs, off, 16);
        Lrow[qt][r] = Lrow[qt][r] * sc + rs;
        Mrow[qt][r] = Mnew;
#pragma unroll
        for (int j = 0; j < 4; ++j) o[qt][j][r] *= sc;
        const int prow = r + hi * 8;
        myP[prow * 32 + nlo]      = (half_t)p0;
        myP[prow * 32 + 16 + nlo] = (half_t)p1;
      }

      // P C-layout -> A-layout via LDS (same wave: DS ops are in-order)
      v16h ap = load_frag(myP + nlo * 32 + hi * 8, myP + nlo * 32 + 16 + hi * 8);
#pragma unroll
      for (int j = 0; j < 4; ++j)
        o[qt][j] = wmma_f16(ap, vf[j], o[qt][j]);
    }
  }

  // normalize and write merged-heads f16 activations
#pragma unroll
  for (int qt = 0; qt < 2; ++qt)
#pragma unroll
    for (int r = 0; r < 8; ++r) {
      const float inv = 1.0f / Lrow[qt][r];
      const size_t grow = (size_t)(b * 2048 + qbase + qt * 16 + r + hi * 8);
#pragma unroll
      for (int j = 0; j < 4; ++j)
        ah[grow * 1024 + h * 64 + j * 16 + nlo] = (half_t)(o[qt][j][r] * inv);
    }
}

// ----------------------------------------------------------------- proj GEMM
// out[4096][1024] f32 = Ah @ w_proj^T + b_proj, 32x64 macro-tile per wave.
__global__ void __launch_bounds__(256) gemm_proj_kernel(
    const half_t* __restrict__ A,    // [4096][1024]
    const half_t* __restrict__ WT,   // [1024][1024] (w_proj transposed)
    const float*  __restrict__ bias, // [1024]
    float* __restrict__ out)         // [4096][1024] f32
{
  const int lane = threadIdx.x & 31;
  const int nlo  = lane & 15;
  const int hi   = lane >> 4;
  const int wave = blockIdx.x * 8 + (threadIdx.x >> 5);
  const int NTile = 1024 / 64;              // 16 macro-col tiles
  const int mt = wave / NTile;              // 0..127
  const int nt = wave - mt * NTile;

  const half_t* aB0 = A + (size_t)(mt * 32 + nlo) * 1024 + hi * 8;
  const half_t* aB1 = aB0 + (size_t)16 * 1024;
  const half_t* bB[4];
#pragma unroll
  for (int j = 0; j < 4; ++j)
    bB[j] = WT + (size_t)(nt * 64 + j * 16 + nlo) * 1024 + hi * 8;

  v8f acc[2][4] = {};
  for (int kk = 0; kk < 1024; kk += 32) {
    __builtin_prefetch(aB0 + kk + 256, 0, 1);
    __builtin_prefetch(bB[0] + kk + 256, 0, 1);
    v16h a0 = load_frag(aB0 + kk, aB0 + kk + 16);
    v16h a1 = load_frag(aB1 + kk, aB1 + kk + 16);
#pragma unroll
    for (int j = 0; j < 4; ++j) {
      v16h bf = load_frag(bB[j] + kk, bB[j] + kk + 16);
      acc[0][j] = wmma_f16(a0, bf, acc[0][j]);
      acc[1][j] = wmma_f16(a1, bf, acc[1][j]);
    }
  }

#pragma unroll
  for (int j = 0; j < 4; ++j) {
    const int n  = nt * 64 + j * 16 + nlo;
    const float bn = bias[n];
#pragma unroll
    for (int i = 0; i < 2; ++i)
#pragma unroll
      for (int r = 0; r < 8; ++r) {
        const int m = mt * 32 + i * 16 + r + hi * 8;
        out[(size_t)m * 1024 + n] = acc[i][j][r] + bn;
      }
  }
}

// -------------------------------------------------------------------- launch
extern "C" void kernel_launch(void* const* d_in, const int* in_sizes, int n_in,
                              void* d_out, int out_size, void* d_ws, size_t ws_size,
                              hipStream_t stream) {
  (void)in_sizes; (void)n_in; (void)out_size; (void)ws_size;
  const float* x      = (const float*)d_in[0];
  // d_in[1] = attention_mask: additive causal mask, applied analytically
  const float* w_attn = (const float*)d_in[2];
  const float* b_attn = (const float*)d_in[3];
  const float* w_proj = (const float*)d_in[4];
  const float* b_proj = (const float*)d_in[5];

  char* ws = (char*)d_ws;
  size_t off = 0;
  auto alloc = [&](size_t bytes) {
    char* p = ws + off;
    off += (bytes + 255) & ~(size_t)255;
    return p;
  };
  half_t* xh  = (half_t*)alloc(4096ull * 1024 * 2);   // x in f16
  half_t* waT = (half_t*)alloc(3072ull * 1024 * 2);   // w_attn^T f16
  half_t* wpT = (half_t*)alloc(1024ull * 1024 * 2);   // w_proj^T f16
  half_t* qk  = (half_t*)alloc(4096ull * 2048 * 2);   // q|k per row
  half_t* vT  = (half_t*)alloc(2048ull * 2048 * 2);   // v[b][h][d][s]
  half_t* ah  = (half_t*)alloc(4096ull * 1024 * 2);   // attention output f16
  // total workspace ~48 MB

  cvt_kernel<<<(4194304 + 255) / 256, 256, 0, stream>>>(x, xh, 4194304);
  cvtT_kernel<<<(3145728 + 255) / 256, 256, 0, stream>>>(w_attn, waT, 1024, 3072);
  cvtT_kernel<<<(1048576 + 255) / 256, 256, 0, stream>>>(w_proj, wpT, 1024, 1024);

  // 128 x 48 = 6144 waves / 8 per block
  gemm_qkv_kernel<<<768, 256, 0, stream>>>(xh, waT, b_attn, qk, vT);
  // 2*16*64 = 2048 waves
  attn_kernel<<<256, 256, 0, stream>>>(qk, vT, ah);
  // 128 x 16 = 2048 waves
  gemm_proj_kernel<<<256, 256, 0, stream>>>(ah, wpT, b_proj, (float*)d_out);
}
